// BackboneDEARI_489626272192
// MI455X (gfx1250) — matrix operand order repaired and checked by reading, not compiled, not run on hardware
//
#include <hip/hip_runtime.h>
#include <hip/hip_bf16.h>

// Problem constants (match reference)
#define Bdim 256
#define Tdim 128
#define Fdim 256
#define Hdim 512

typedef __attribute__((ext_vector_type(16))) __bf16 v16bf;
typedef __attribute__((ext_vector_type(8)))  __bf16 v8bf;
typedef __attribute__((ext_vector_type(8)))  float  v8f;

__device__ __forceinline__ float sigmoidf_(float x) { return 1.0f / (1.0f + __expf(-x)); }

// ---------------------------------------------------------------------------
// WMMA fragment loaders (16x16x32 bf16, wave32; A/B use contiguous k = 16*half + elem).
// All operands pre-converted to bf16, so the inner loop has no cvt VALU work.
// A (16x32): lanes 0-15 : M=lane,    elems [0..15]=K 0..15
//            lanes 16-31: M=lane-16, elems [0..15]=K 16..31
// ---------------------------------------------------------------------------
__device__ __forceinline__ v16bf load_a_frag(const __bf16* A, size_t lda, int m0, int kb) {
    int lane = threadIdx.x & 31;
    int half = lane >> 4;
    int m    = m0 + (lane & 15);
    const __bf16* p = A + (size_t)m * lda + (size_t)(kb + half * 16);
    v8bf lo = *(const v8bf*)p;          // 16B aligned
    v8bf hi = *(const v8bf*)(p + 8);
    v16bf f;
#pragma unroll
    for (int i = 0; i < 8; ++i) { f[i] = lo[i]; f[i + 8] = hi[i]; }
    return f;
}

// B (32x16) from row-major weight W[N,K] (bf16):
//   lanes 0-15 : N=lane,    elems [0..15] = K 0..15  (contiguous in W row)
//   lanes 16-31: N=lane-16, elems [0..15] = K 16..31
__device__ __forceinline__ v16bf load_b_frag(const __bf16* W, size_t ldw, int n0, int kb) {
    int lane = threadIdx.x & 31;
    int n    = n0 + (lane & 15);
    const __bf16* p = W + (size_t)n * ldw + (size_t)(kb + ((lane >> 4) << 4));
    return *(const v16bf*)p;            // 32B aligned
}

// Register-blocked wave GEMM: 32 x (16*NB) output tile per wave.
// Per K-step: 2 A-frags + NB B-frags -> 2*NB WMMAs.
template <int NB>
__device__ __forceinline__ void gemm_block(const __bf16* A, size_t lda,
                                           const __bf16* W, size_t ldw,
                                           int m0, int n0, int K, v8f (&acc)[2][NB]) {
    for (int kb = 0; kb < K; kb += 32) {
        v16bf a0 = load_a_frag(A, lda, m0, kb);
        v16bf a1 = load_a_frag(A, lda, m0 + 16, kb);
#pragma unroll
        for (int j = 0; j < NB; ++j) {
            v16bf b = load_b_frag(W, ldw, n0 + 16 * j, kb);
            acc[0][j] = __builtin_amdgcn_wmma_f32_16x16x32_bf16(false, a0, false, b,
                                                                (short)0, acc[0][j], false, false);
            acc[1][j] = __builtin_amdgcn_wmma_f32_16x16x32_bf16(false, a1, false, b,
                                                                (short)0, acc[1][j], false, false);
        }
    }
}

// C/D layout: VGPR i -> lanes 0-15: (M=i, N=lane); lanes 16-31: (M=8+i, N=lane-16)
// Wave tile: m0 += 32 per M step, n0 += 16*NB per N step.
#define WAVE_TILE(NB, tiles_n)                                        \
    int wave = (int)((blockIdx.x * blockDim.x + threadIdx.x) >> 5);   \
    int m0 = (wave / (tiles_n)) * 32;                                 \
    int n0 = (wave % (tiles_n)) * (16 * (NB));                        \
    int lane = threadIdx.x & 31;                                      \
    int nlo  = lane & 15;                                             \
    int mho  = (lane >> 4) << 3;

// ---------------------------------------------------------------------------
// Setup kernels
// ---------------------------------------------------------------------------
__global__ void zero_f32_kernel(float* p, int n) {
    int i = blockIdx.x * blockDim.x + threadIdx.x;
    if (i < n) p[i] = 0.0f;
}

__global__ void cvt_bf16_kernel(const float* __restrict__ src, __bf16* __restrict__ dst,
                                int n, int ld, int zero_diag) {
    int i = blockIdx.x * blockDim.x + threadIdx.x;
    if (i < n) {
        float v = src[i];
        if (zero_diag && (i / ld) == (i % ld)) v = 0.0f;
        dst[i] = (__bf16)v;
    }
}

// ---------------------------------------------------------------------------
// Step kernel 1 (elementwise, runs first): gamma_x; comb_bf=[gamma_x, m];
// rnn_bf[:,F:]=m; d_bf = bf16(deltas_t)
// ---------------------------------------------------------------------------
__global__ void prestep_kernel(const float* __restrict__ deltas_t,
                               const float* __restrict__ mask_t,
                               const float* __restrict__ wgx,
                               const float* __restrict__ bgx,
                               __bf16* __restrict__ comb_bf,
                               __bf16* __restrict__ rnn_bf,
                               __bf16* __restrict__ d_bf) {
    int idx = blockIdx.x * blockDim.x + threadIdx.x;  // B*F
    int b = idx >> 8, f = idx & (Fdim - 1);
    size_t gix = (size_t)b * (Tdim * Fdim) + f;
    float d = deltas_t[gix];
    float m = mask_t[gix];
    float gx = __expf(-fmaxf(fmaf(d, wgx[f], bgx[f]), 0.0f));
    comb_bf[(size_t)b * (2 * Fdim) + f]        = (__bf16)gx;
    comb_bf[(size_t)b * (2 * Fdim) + Fdim + f] = (__bf16)m;
    rnn_bf [(size_t)b * (2 * Fdim) + Fdim + f] = (__bf16)m;
    d_bf   [(size_t)b * Fdim + f]              = (__bf16)d;
}

// ---------------------------------------------------------------------------
// Step kernel 2: gamma_h = exp(-relu(d_bf @ W_gh^T + b_gh)); h *= gamma_h;
//                h_bf = bf16(h)   (consumed by hist + gates GEMMs)
// ---------------------------------------------------------------------------
__global__ void decay_gemm_kernel(const __bf16* __restrict__ d_bf,
                                  const __bf16* __restrict__ Wgh,
                                  const float* __restrict__ bgh,
                                  float* __restrict__ h,
                                  __bf16* __restrict__ h_bf) {
    WAVE_TILE(2, Hdim / 32);
    v8f acc[2][2] = {};
    gemm_block<2>(d_bf, Fdim, Wgh, Fdim, m0, n0, Fdim, acc);
#pragma unroll
    for (int mi = 0; mi < 2; ++mi)
#pragma unroll
        for (int nj = 0; nj < 2; ++nj) {
            int col = n0 + 16 * nj + nlo;
            float bias = bgh[col];
#pragma unroll
            for (int i = 0; i < 8; ++i) {
                int row = m0 + 16 * mi + mho + i;
                size_t hix = (size_t)row * Hdim + col;
                float g  = __expf(-fmaxf(acc[mi][nj][i] + bias, 0.0f));
                float hv = h[hix] * g;
                h[hix]    = hv;
                h_bf[hix] = (__bf16)hv;
            }
        }
}

// ---------------------------------------------------------------------------
// Step kernel 3: x_h = h @ W_hist^T + b; xr_bf = bf16(m*x + (1-m)*x_h)
// ---------------------------------------------------------------------------
__global__ void hist_gemm_kernel(const __bf16* __restrict__ h_bf,
                                 const __bf16* __restrict__ Whist,
                                 const float* __restrict__ bhist,
                                 const float* __restrict__ x_t,
                                 const float* __restrict__ mask_t,
                                 float* __restrict__ xh,
                                 __bf16* __restrict__ xr_bf) {
    WAVE_TILE(2, Fdim / 32);
    v8f acc[2][2] = {};
    gemm_block<2>(h_bf, Hdim, Whist, Hdim, m0, n0, Hdim, acc);
#pragma unroll
    for (int mi = 0; mi < 2; ++mi)
#pragma unroll
        for (int nj = 0; nj < 2; ++nj) {
            int col = n0 + 16 * nj + nlo;
            float bias = bhist[col];
#pragma unroll
            for (int i = 0; i < 8; ++i) {
                int row = m0 + 16 * mi + mho + i;
                size_t ix  = (size_t)row * Fdim + col;
                size_t gix = (size_t)row * (Tdim * Fdim) + col;
                float xhv = acc[mi][nj][i] + bias;
                float mv  = mask_t[gix];
                xh[ix]    = xhv;
                xr_bf[ix] = (__bf16)(mv * x_t[gix] + (1.0f - mv) * xhv);
            }
        }
}

// ---------------------------------------------------------------------------
// Step kernel 4: xu = x_r @ W_fr^T + b_fr  (diag zeroed in bf16 W_fr)
// ---------------------------------------------------------------------------
__global__ void fr_gemm_kernel(const __bf16* __restrict__ xr_bf,
                               const __bf16* __restrict__ Wfr,
                               const float* __restrict__ bfr,
                               float* __restrict__ xu) {
    WAVE_TILE(2, Fdim / 32);
    v8f acc[2][2] = {};
    gemm_block<2>(xr_bf, Fdim, Wfr, Fdim, m0, n0, Fdim, acc);
#pragma unroll
    for (int mi = 0; mi < 2; ++mi)
#pragma unroll
        for (int nj = 0; nj < 2; ++nj) {
            int col = n0 + 16 * nj + nlo;
            float bias = bfr[col];
#pragma unroll
            for (int i = 0; i < 8; ++i)
                xu[(size_t)(m0 + 16 * mi + mho + i) * Fdim + col] = acc[mi][nj][i] + bias;
        }
}

// ---------------------------------------------------------------------------
// Step kernel 5: beta = sigmoid(comb_bf @ W_comb^T + b); x_comb; loss; x_imp
// ---------------------------------------------------------------------------
__global__ void comb_gemm_kernel(const __bf16* __restrict__ comb_bf,
                                 const __bf16* __restrict__ Wcomb,
                                 const float* __restrict__ bcomb,
                                 const float* __restrict__ xh,
                                 const float* __restrict__ xu,
                                 const float* __restrict__ x_t,
                                 const float* __restrict__ mask_t,
                                 float* __restrict__ recon_t,
                                 float* __restrict__ ximp_t,
                                 __bf16* __restrict__ rnn_bf,
                                 float* __restrict__ num_acc,
                                 float* __restrict__ den_acc) {
    WAVE_TILE(2, Fdim / 32);
    v8f acc[2][2] = {};
    gemm_block<2>(comb_bf, 2 * Fdim, Wcomb, 2 * Fdim, m0, n0, 2 * Fdim, acc);
    float lnum = 0.0f, lden = 0.0f;
#pragma unroll
    for (int mi = 0; mi < 2; ++mi)
#pragma unroll
        for (int nj = 0; nj < 2; ++nj) {
            int col = n0 + 16 * nj + nlo;
            float bias = bcomb[col];
#pragma unroll
            for (int i = 0; i < 8; ++i) {
                int row = m0 + 16 * mi + mho + i;
                size_t ix  = (size_t)row * Fdim + col;
                size_t gix = (size_t)row * (Tdim * Fdim) + col;
                float beta = sigmoidf_(acc[mi][nj][i] + bias);
                float xc   = beta * xu[ix] + (1.0f - beta) * xh[ix];
                float xv   = x_t[gix];
                float mv   = mask_t[gix];
                recon_t[gix] = xc;
                float xi = mv * xv + (1.0f - mv) * xc;
                ximp_t[gix] = xi;
                rnn_bf[(size_t)row * (2 * Fdim) + col] = (__bf16)xi;
                lnum += fabsf(xc - xv) * mv;
                lden += mv;
            }
        }
#pragma unroll
    for (int off = 16; off > 0; off >>= 1) {
        lnum += __shfl_xor(lnum, off, 32);
        lden += __shfl_xor(lden, off, 32);
    }
    if ((threadIdx.x & 31) == 0) {
        atomicAdd(num_acc, lnum);
        atomicAdd(den_acc, lden);
    }
}

// ---------------------------------------------------------------------------
// Step kernel 6: gates = rnn_bf @ W_ih^T + h_bf @ W_hh^T + b_ih + b_hh
// (largest GEMM: 256x1024x2048 -> wave tile 32x64, NB=4)
// ---------------------------------------------------------------------------
__global__ void gates_gemm_kernel(const __bf16* __restrict__ rnn_bf,
                                  const __bf16* __restrict__ Wih,
                                  const float* __restrict__ bih,
                                  const __bf16* __restrict__ h_bf,
                                  const __bf16* __restrict__ Whh,
                                  const float* __restrict__ bhh,
                                  float* __restrict__ gates) {
    WAVE_TILE(4, (4 * Hdim) / 64);
    v8f acc[2][4] = {};
    gemm_block<4>(rnn_bf, 2 * Fdim, Wih, 2 * Fdim, m0, n0, 2 * Fdim, acc);
    gemm_block<4>(h_bf,   Hdim,     Whh, Hdim,     m0, n0, Hdim,     acc);
#pragma unroll
    for (int mi = 0; mi < 2; ++mi)
#pragma unroll
        for (int nj = 0; nj < 4; ++nj) {
            int col = n0 + 16 * nj + nlo;
            float bias = bih[col] + bhh[col];
#pragma unroll
            for (int i = 0; i < 8; ++i)
                gates[(size_t)(m0 + 16 * mi + mho + i) * (4 * Hdim) + col] =
                    acc[mi][nj][i] + bias;
        }
}

// ---------------------------------------------------------------------------
// Step kernel 7: LSTM pointwise update
// ---------------------------------------------------------------------------
__global__ void lstm_kernel(const float* __restrict__ gates,
                            float* __restrict__ c, float* __restrict__ h) {
    int idx = blockIdx.x * blockDim.x + threadIdx.x;  // B*H
    int b = idx >> 9, j = idx & (Hdim - 1);
    const float* g = gates + (size_t)b * (4 * Hdim);
    float ig = sigmoidf_(g[j]);
    float fg = sigmoidf_(g[Hdim + j]);
    float gg = tanhf(g[2 * Hdim + j]);
    float og = sigmoidf_(g[3 * Hdim + j]);
    float cv = fg * c[idx] + ig * gg;
    c[idx] = cv;
    h[idx] = og * tanhf(cv);
}

// ---------------------------------------------------------------------------
// Finalize: h -> out, x_loss = sum_t num[t]/(den[t]+1e-12), kl = 0
// ---------------------------------------------------------------------------
__global__ void copy_h_kernel(const float* __restrict__ h, float* __restrict__ out) {
    int i = blockIdx.x * blockDim.x + threadIdx.x;
    out[i] = h[i];
}

__global__ void finalize_loss_kernel(const float* __restrict__ num,
                                     const float* __restrict__ den,
                                     float* __restrict__ out_loss) {
    if (threadIdx.x == 0 && blockIdx.x == 0) {
        float loss = 0.0f;
        for (int t = 0; t < Tdim; ++t) loss += num[t] / (den[t] + 1e-12f);
        out_loss[0] = loss;  // x_loss
        out_loss[1] = 0.0f;  // kl
    }
}

// ---------------------------------------------------------------------------
// Host launcher
// ---------------------------------------------------------------------------
extern "C" void kernel_launch(void* const* d_in, const int* in_sizes, int n_in,
                              void* d_out, int out_size, void* d_ws, size_t ws_size,
                              hipStream_t stream) {
    (void)in_sizes; (void)n_in; (void)out_size; (void)ws_size;

    const float* x      = (const float*)d_in[0];
    const float* mask   = (const float*)d_in[1];
    const float* deltas = (const float*)d_in[2];
    const float* W_gh   = (const float*)d_in[3];
    const float* b_gh   = (const float*)d_in[4];
    const float* w_gx   = (const float*)d_in[5];
    const float* b_gx   = (const float*)d_in[6];
    const float* W_hist = (const float*)d_in[7];
    const float* b_hist = (const float*)d_in[8];
    const float* W_fr   = (const float*)d_in[9];
    const float* b_fr   = (const float*)d_in[10];
    const float* W_comb = (const float*)d_in[11];
    const float* b_comb = (const float*)d_in[12];
    const float* W_ih   = (const float*)d_in[13];
    const float* b_ih   = (const float*)d_in[14];
    const float* W_hh   = (const float*)d_in[15];
    const float* b_hh   = (const float*)d_in[16];

    float* out = (float*)d_out;
    float* ximp_out  = out;                                     // [B,T,F]
    float* recon_out = out + (size_t)Bdim * Tdim * Fdim;        // [B,T,F]
    float* h_out     = out + 2 * (size_t)Bdim * Tdim * Fdim;    // [B,H]
    float* loss_out  = h_out + (size_t)Bdim * Hdim;             // [x_loss, kl]

    // Workspace bump allocation: fp32 region, then bf16 weights + activations
    float* wsf = (float*)d_ws;
    size_t o = 0;
    float* ws_h     = wsf + o; o += (size_t)Bdim * Hdim;         // 131072
    float* ws_c     = wsf + o; o += (size_t)Bdim * Hdim;         // 131072
    float* ws_xh    = wsf + o; o += (size_t)Bdim * Fdim;         // 65536
    float* ws_xu    = wsf + o; o += (size_t)Bdim * Fdim;         // 65536
    float* ws_gates = wsf + o; o += (size_t)Bdim * 4 * Hdim;     // 524288
    float* ws_num   = wsf + o; o += Tdim;
    float* ws_den   = wsf + o; o += Tdim;
    __bf16* bfp = (__bf16*)(wsf + o);
    __bf16* Wgh_bf   = bfp; bfp += (size_t)Hdim * Fdim;
    __bf16* Whist_bf = bfp; bfp += (size_t)Fdim * Hdim;
    __bf16* Wfr_bf   = bfp; bfp += (size_t)Fdim * Fdim;
    __bf16* Wcomb_bf = bfp; bfp += (size_t)Fdim * 2 * Fdim;
    __bf16* Wih_bf   = bfp; bfp += (size_t)4 * Hdim * 2 * Fdim;
    __bf16* Whh_bf   = bfp; bfp += (size_t)4 * Hdim * Hdim;
    __bf16* h_bf     = bfp; bfp += (size_t)Bdim * Hdim;
    __bf16* d_bf     = bfp; bfp += (size_t)Bdim * Fdim;
    __bf16* xr_bf    = bfp; bfp += (size_t)Bdim * Fdim;
    __bf16* comb_bf  = bfp; bfp += (size_t)Bdim * 2 * Fdim;
    __bf16* rnn_bf   = bfp; bfp += (size_t)Bdim * 2 * Fdim;

    const int TB = 256;

    // --- init: zero h, c (contiguous) and num/den (contiguous) ---
    zero_f32_kernel<<<(2 * Bdim * Hdim + TB - 1) / TB, TB, 0, stream>>>(ws_h, 2 * Bdim * Hdim);
    zero_f32_kernel<<<1, TB, 0, stream>>>(ws_num, 2 * Tdim);

    // --- one-time fp32 -> bf16 weight conversion (W_fr with zeroed diagonal) ---
    cvt_bf16_kernel<<<(Hdim * Fdim + TB - 1) / TB, TB, 0, stream>>>(W_gh, Wgh_bf, Hdim * Fdim, Fdim, 0);
    cvt_bf16_kernel<<<(Fdim * Hdim + TB - 1) / TB, TB, 0, stream>>>(W_hist, Whist_bf, Fdim * Hdim, Hdim, 0);
    cvt_bf16_kernel<<<(Fdim * Fdim + TB - 1) / TB, TB, 0, stream>>>(W_fr, Wfr_bf, Fdim * Fdim, Fdim, 1);
    cvt_bf16_kernel<<<(Fdim * 2 * Fdim + TB - 1) / TB, TB, 0, stream>>>(W_comb, Wcomb_bf, Fdim * 2 * Fdim, 2 * Fdim, 0);
    cvt_bf16_kernel<<<(4 * Hdim * 2 * Fdim + TB - 1) / TB, TB, 0, stream>>>(W_ih, Wih_bf, 4 * Hdim * 2 * Fdim, 2 * Fdim, 0);
    cvt_bf16_kernel<<<(4 * Hdim * Hdim + TB - 1) / TB, TB, 0, stream>>>(W_hh, Whh_bf, 4 * Hdim * Hdim, Hdim, 0);

    // Wave-tile grids: 8 waves (256 threads) per block, exact tile counts
    const int gDecay = (Bdim / 32) * (Hdim / 32) / 8;        // 16 blocks (NB=2)
    const int gF     = (Bdim / 32) * (Fdim / 32) / 8;        // 8 blocks  (NB=2)
    const int gGates = (Bdim / 32) * (4 * Hdim / 64) / 8;    // 32 blocks (NB=4)

    for (int t = 0; t < Tdim; ++t) {
        const float* x_t = x      + (size_t)t * Fdim;
        const float* m_t = mask   + (size_t)t * Fdim;
        const float* d_t = deltas + (size_t)t * Fdim;
        float* recon_t = recon_out + (size_t)t * Fdim;
        float* ximp_t  = ximp_out  + (size_t)t * Fdim;

        prestep_kernel<<<Bdim * Fdim / TB, TB, 0, stream>>>(d_t, m_t, w_gx, b_gx,
                                                            comb_bf, rnn_bf, d_bf);
        decay_gemm_kernel<<<gDecay, TB, 0, stream>>>(d_bf, Wgh_bf, b_gh, ws_h, h_bf);
        hist_gemm_kernel<<<gF, TB, 0, stream>>>(h_bf, Whist_bf, b_hist, x_t, m_t,
                                                ws_xh, xr_bf);
        fr_gemm_kernel<<<gF, TB, 0, stream>>>(xr_bf, Wfr_bf, b_fr, ws_xu);
        comb_gemm_kernel<<<gF, TB, 0, stream>>>(comb_bf, Wcomb_bf, b_comb,
                                                ws_xh, ws_xu, x_t, m_t,
                                                recon_t, ximp_t, rnn_bf,
                                                ws_num + t, ws_den + t);
        gates_gemm_kernel<<<gGates, TB, 0, stream>>>(rnn_bf, Wih_bf, b_ih,
                                                     h_bf, Whh_bf, b_hh, ws_gates);
        lstm_kernel<<<Bdim * Hdim / TB, TB, 0, stream>>>(ws_gates, ws_c, ws_h);
    }

    copy_h_kernel<<<Bdim * Hdim / TB, TB, 0, stream>>>(ws_h, h_out);
    finalize_loss_kernel<<<1, 32, 0, stream>>>(ws_num, ws_den, loss_out);
}
